// EdgeBiasedAttention_14181982012070
// MI455X (gfx1250) — compile-verified, matrix operand order checked by reading, not run on hardware
//
#include <hip/hip_runtime.h>
#include <math.h>

// EdgeBiasedAttention for MI455X (gfx1250, wave32, WMMA).
// Pipeline: [A] edge bias (1GB-stream bf16-WMMA GEMM + fused LN/mean),
//           [B] QKV projection, [C] per-(head,qtile) attention, [D] out proj.

typedef __attribute__((ext_vector_type(16))) __bf16 v16bf;
typedef __attribute__((ext_vector_type(8)))  float  v8f;

#define WMMA_BF16(A, B, C) \
  __builtin_amdgcn_wmma_f32_16x16x32_bf16(false, (A), false, (B), (short)0, (C), false, false)

constexpr int NSEQ = 1024;
constexpr int DIMC = 128;
constexpr int HEADS = 8;
constexpr int HD = 16;

// ---------------------------------------------------------------------------
// Kernel A: edge bias.  h = ef@W + eb  (per 1M edge rows, 128 channels)
// bias = inv_sigma*(mean(h*g) - mu*mean(g)) + mean(b)   (LN+mean fused)
// ---------------------------------------------------------------------------
__global__ __launch_bounds__(256) void edge_bias_kernel(
    const float* __restrict__ ef,   // [N*N, 256]
    const float* __restrict__ ew,   // [256, 128]
    const float* __restrict__ eb,   // [128]
    const float* __restrict__ lng,  // [128]
    const float* __restrict__ lnb,  // [128]
    float* __restrict__ bias)       // [N*N]
{
  __shared__ __bf16 Wt[DIMC * 256];   // col-major [c][k], 64 KB

  const int tid = threadIdx.x;
  for (int i = tid; i < DIMC * 256; i += 256) {
    int k = i >> 7, c = i & 127;
    Wt[c * 256 + k] = (__bf16)ew[i];
  }
  float gs = 0.f, bs = 0.f;
  for (int i = 0; i < DIMC; ++i) { gs += lng[i]; bs += lnb[i]; }
  const float gmean = gs * (1.f / 128.f), bmean = bs * (1.f / 128.f);
  __syncthreads();

  const int lane = tid & 31, wave = tid >> 5;
  const int lrow = lane & 15;
  const int half = lane >> 4;
  const int k0a  = half * 8;

  float ebv[8], gv[8];
#pragma unroll
  for (int nt = 0; nt < 8; ++nt) {
    int col = nt * 16 + lrow;
    ebv[nt] = eb[col];
    gv[nt]  = lng[col];
  }

  const int gwave  = blockIdx.x * 8 + wave;
  const int nwave  = gridDim.x * 8;
  const int ntiles = (NSEQ * NSEQ) / 16;

  for (int mt = gwave; mt < ntiles; mt += nwave) {
    const float* arow = ef + (size_t)(mt * 16 + lrow) * 256;
    v16bf af[8];
#pragma unroll
    for (int kt = 0; kt < 8; ++kt) {
      const float* p = arow + kt * 32 + k0a;
#pragma unroll
      for (int i = 0; i < 8; ++i) {
        af[kt][i]     = (__bf16)p[i];
        af[kt][i + 8] = (__bf16)p[i + 16];
      }
    }
    float s1[8], sg[8], s2[8];
#pragma unroll
    for (int r = 0; r < 8; ++r) { s1[r] = 0.f; sg[r] = 0.f; s2[r] = 0.f; }

#pragma unroll
    for (int nt = 0; nt < 8; ++nt) {
      int col = nt * 16 + lrow;
      v8f acc = {0.f, 0.f, 0.f, 0.f, 0.f, 0.f, 0.f, 0.f};
#pragma unroll
      for (int kt = 0; kt < 8; ++kt) {
        const __bf16* wp = &Wt[col * 256 + kt * 32 + k0a];
        v16bf bfr;
#pragma unroll
        for (int i = 0; i < 8; ++i) { bfr[i] = wp[i]; bfr[i + 8] = wp[i + 16]; }
        acc = WMMA_BF16(af[kt], bfr, acc);
      }
#pragma unroll
      for (int r = 0; r < 8; ++r) {
        float h = acc[r] + ebv[nt];
        s1[r] += h; sg[r] += h * gv[nt]; s2[r] += h * h;
      }
    }
#pragma unroll
    for (int m = 1; m < 16; m <<= 1) {
#pragma unroll
      for (int r = 0; r < 8; ++r) {
        s1[r] += __shfl_xor(s1[r], m, 32);
        sg[r] += __shfl_xor(sg[r], m, 32);
        s2[r] += __shfl_xor(s2[r], m, 32);
      }
    }
    if (lrow == 0) {
#pragma unroll
      for (int r = 0; r < 8; ++r) {
        float mu  = s1[r] * (1.f / 128.f);
        float var = s2[r] * (1.f / 128.f) - mu * mu;
        float inv = rsqrtf(var + 1e-5f);
        bias[mt * 16 + half * 8 + r] =
            inv * (sg[r] * (1.f / 128.f) - mu * gmean) + bmean;
      }
    }
  }
}

// ---------------------------------------------------------------------------
// Kernel B: QKV = x @ qkv_w + qkv_b, scattered into [3][H][N][HD] fp32.
// ---------------------------------------------------------------------------
__global__ __launch_bounds__(256) void qkv_kernel(
    const float* __restrict__ x, const float* __restrict__ w,
    const float* __restrict__ b, float* __restrict__ qkvws)
{
  const int tid = threadIdx.x, lane = tid & 31, wave = tid >> 5;
  const int gw = blockIdx.x * 8 + wave;     // 0..1535
  const int mt = gw / 24, nt = gw % 24;
  const int lrow = lane & 15, half = lane >> 4, k0a = half * 8;
  const int col = nt * 16 + lrow;
  const float* arow = x + (size_t)(mt * 16 + lrow) * DIMC;

  v8f acc = {0.f, 0.f, 0.f, 0.f, 0.f, 0.f, 0.f, 0.f};
#pragma unroll
  for (int kt = 0; kt < 4; ++kt) {
    v16bf af, bfr;
    const float* ap = arow + kt * 32 + k0a;
#pragma unroll
    for (int i = 0; i < 8; ++i) {
      af[i]      = (__bf16)ap[i];
      af[i + 8]  = (__bf16)ap[i + 16];
      bfr[i]     = (__bf16)w[(size_t)(kt * 32 + k0a + i) * 384 + col];
      bfr[i + 8] = (__bf16)w[(size_t)(kt * 32 + k0a + 16 + i) * 384 + col];
    }
    acc = WMMA_BF16(af, bfr, acc);
  }
  const float bb = b[col];
  const int which = col >> 7, hh = (col >> 4) & 7, d = col & 15;
  float* dst = qkvws + (size_t)(which * HEADS + hh) * NSEQ * HD;
#pragma unroll
  for (int r = 0; r < 8; ++r) {
    int row = mt * 16 + half * 8 + r;
    dst[row * HD + d] = acc[r] + bb;
  }
}

// ---------------------------------------------------------------------------
// Kernel C: one (head, q-tile) per block. S=QK^T*scale+bias -> softmax -> PV.
// ---------------------------------------------------------------------------
__global__ __launch_bounds__(256) void attn_kernel(
    const float* __restrict__ qws, const float* __restrict__ kws,
    const float* __restrict__ vws, const float* __restrict__ bias,
    float* __restrict__ attn_out)
{
  __shared__ float sc[NSEQ * 16];   // [key][q], 64 KB (reused as reduce buf)
  const int h  = blockIdx.x >> 6;
  const int qt = blockIdx.x & 63;
  const int tid = threadIdx.x, lane = tid & 31, wave = tid >> 5;
  const int lrow = lane & 15, half = lane >> 4, k0a = half * 8;
  const float* Qh = qws + (size_t)h * NSEQ * HD;
  const float* Kh = kws + (size_t)h * NSEQ * HD;
  const float* Vh = vws + (size_t)h * NSEQ * HD;

  // phase 1: scores (head_dim=16 zero-padded to K=32)
  v16bf qf;
  {
    const float* qp = Qh + (size_t)(qt * 16 + lrow) * HD + k0a;
#pragma unroll
    for (int i = 0; i < 8; ++i) { qf[i] = (__bf16)qp[i]; qf[i + 8] = (__bf16)0.f; }
  }
#pragma unroll
  for (int ii = 0; ii < 8; ++ii) {
    int kt = wave * 8 + ii;
    int key = kt * 16 + lrow;
    v16bf kf;
    const float* kp = Kh + (size_t)key * HD + k0a;
#pragma unroll
    for (int i = 0; i < 8; ++i) { kf[i] = (__bf16)kp[i]; kf[i + 8] = (__bf16)0.f; }
    v8f acc = {0.f, 0.f, 0.f, 0.f, 0.f, 0.f, 0.f, 0.f};
    acc = WMMA_BF16(qf, kf, acc);
#pragma unroll
    for (int r = 0; r < 8; ++r) {
      int rg = half * 8 + r;
      sc[key * 16 + rg] =
          acc[r] * 0.25f + bias[(size_t)(qt * 16 + rg) * NSEQ + key];
    }
  }
  __syncthreads();

  // phase 2: softmax per q row (16 threads/row, shuffle reduce)
  const int r2 = tid >> 4, c0 = tid & 15;
  float mx = -3.0e38f;
  for (int i = 0; i < 64; ++i) mx = fmaxf(mx, sc[(c0 + 16 * i) * 16 + r2]);
#pragma unroll
  for (int m = 1; m < 16; m <<= 1) mx = fmaxf(mx, __shfl_xor(mx, m, 32));
  float sum = 0.f;
  for (int i = 0; i < 64; ++i) {
    int idx = (c0 + 16 * i) * 16 + r2;
    float p = __expf(sc[idx] - mx);
    sc[idx] = p;
    sum += p;
  }
#pragma unroll
  for (int m = 1; m < 16; m <<= 1) sum += __shfl_xor(sum, m, 32);
  const float inv_sum = 1.f / sum;
  __syncthreads();

  // phase 3: O = P @ V, each wave covers 4 chunks of 32 keys
  v8f acc = {0.f, 0.f, 0.f, 0.f, 0.f, 0.f, 0.f, 0.f};
#pragma unroll
  for (int cc = 0; cc < 4; ++cc) {
    int kbase = (wave + cc * 8) * 32 + k0a;
    v16bf pa, vb;
#pragma unroll
    for (int i = 0; i < 8; ++i) {
      pa[i]     = (__bf16)sc[(kbase + i) * 16 + lrow];
      pa[i + 8] = (__bf16)sc[(kbase + 16 + i) * 16 + lrow];
      vb[i]     = (__bf16)Vh[(size_t)(kbase + i) * HD + lrow];
      vb[i + 8] = (__bf16)Vh[(size_t)(kbase + 16 + i) * HD + lrow];
    }
    acc = WMMA_BF16(pa, vb, acc);
  }
  __syncthreads();                      // all sc reads done; reuse as reducer
#pragma unroll
  for (int r = 0; r < 8; ++r)
    sc[wave * 256 + (half * 8 + r) * 16 + lrow] = acc[r];
  __syncthreads();
  {
    float s = 0.f;
#pragma unroll
    for (int w = 0; w < 8; ++w) s += sc[w * 256 + tid];
    attn_out[(size_t)(qt * 16 + r2) * DIMC + h * HD + c0] = s * inv_sum;
  }
}

// ---------------------------------------------------------------------------
// Kernel D: out = attn @ proj_w + proj_b
// ---------------------------------------------------------------------------
__global__ __launch_bounds__(256) void proj_kernel(
    const float* __restrict__ a, const float* __restrict__ w,
    const float* __restrict__ b, float* __restrict__ out)
{
  const int tid = threadIdx.x, lane = tid & 31, wave = tid >> 5;
  const int gw = blockIdx.x * 8 + wave;     // 0..511
  const int mt = gw >> 3, nt = gw & 7;
  const int lrow = lane & 15, half = lane >> 4, k0a = half * 8;
  const int col = nt * 16 + lrow;
  const float* arow = a + (size_t)(mt * 16 + lrow) * DIMC;

  v8f acc = {0.f, 0.f, 0.f, 0.f, 0.f, 0.f, 0.f, 0.f};
#pragma unroll
  for (int kt = 0; kt < 4; ++kt) {
    v16bf af, bfr;
    const float* ap = arow + kt * 32 + k0a;
#pragma unroll
    for (int i = 0; i < 8; ++i) {
      af[i]      = (__bf16)ap[i];
      af[i + 8]  = (__bf16)ap[i + 16];
      bfr[i]     = (__bf16)w[(size_t)(kt * 32 + k0a + i) * DIMC + col];
      bfr[i + 8] = (__bf16)w[(size_t)(kt * 32 + k0a + 16 + i) * DIMC + col];
    }
    acc = WMMA_BF16(af, bfr, acc);
  }
  const float bb = b[col];
#pragma unroll
  for (int r = 0; r < 8; ++r) {
    int row = mt * 16 + half * 8 + r;
    out[(size_t)row * DIMC + col] = acc[r] + bb;
  }
}

// ---------------------------------------------------------------------------
extern "C" void kernel_launch(void* const* d_in, const int* in_sizes, int n_in,
                              void* d_out, int out_size, void* d_ws, size_t ws_size,
                              hipStream_t stream) {
  (void)in_sizes; (void)n_in; (void)out_size; (void)ws_size;
  const float* x    = (const float*)d_in[0];
  const float* ef   = (const float*)d_in[1];
  const float* qkvw = (const float*)d_in[2];
  const float* qkvb = (const float*)d_in[3];
  const float* pw   = (const float*)d_in[4];
  const float* pb   = (const float*)d_in[5];
  const float* ew   = (const float*)d_in[6];
  const float* eb   = (const float*)d_in[7];
  const float* lng  = (const float*)d_in[8];
  const float* lnb  = (const float*)d_in[9];
  float* out = (float*)d_out;

  char* ws = (char*)d_ws;
  float* bias   = (float*)(ws);                  // 4 MB   [1024*1024]
  float* qkvbuf = (float*)(ws + 4194304);        // 1.5 MB [3][8][1024][16]
  float* qb = qkvbuf;
  float* kb = qkvbuf + (size_t)HEADS * NSEQ * HD;
  float* vb = qkvbuf + (size_t)2 * HEADS * NSEQ * HD;
  float* attnb  = (float*)(ws + 5767168);        // 512 KB [1024][128]

  edge_bias_kernel<<<1024, 256, 0, stream>>>(ef, ew, eb, lng, lnb, bias);
  qkv_kernel<<<192, 256, 0, stream>>>(x, qkvw, qkvb, qkvbuf);
  attn_kernel<<<512, 256, 0, stream>>>(qb, kb, vb, bias, attnb);
  proj_kernel<<<64, 256, 0, stream>>>(attnb, pw, pb, out);
}
